// MyTransformer_52089363366218
// MI455X (gfx1250) — matrix operand order repaired and checked by reading, compiled, not run on hardware
//
#include <hip/hip_runtime.h>
#include <hip/hip_bf16.h>
#include <cstdint>

typedef __attribute__((ext_vector_type(16))) __bf16 v16bf;
typedef __attribute__((ext_vector_type(8)))  float  v8f;

#define T_SEQ  1024
#define DMODEL 1024
#define NHEAD  16
#define NGROUP 4
#define HEADD  64
#define FFDIM  4096
#define NLAYER 12
#define VOCAB  32000
#define KVD    (NGROUP * HEADD)   // 256

// f32 -> bf16 RNE via integer math
__device__ __forceinline__ unsigned short f2bfu(float f) {
  unsigned u = __builtin_bit_cast(unsigned, f);
  unsigned r = (u + 0x7fffu + ((u >> 16) & 1u)) >> 16;
  return (unsigned short)r;
}
__device__ __forceinline__ __bf16 f2bf(float f) {
  unsigned short s = f2bfu(f);
  return __builtin_bit_cast(__bf16, s);
}
__device__ __forceinline__ __bf16 u2bf(unsigned short s) {
  return __builtin_bit_cast(__bf16, s);
}

struct alignas(8) bf16x4 { __bf16 v[4]; };

// async raw 16B copy global -> LDS (ASYNCcnt path)
__device__ __forceinline__ void async_copy_b128(const void* gp, void* lp) {
  unsigned lds_off = (unsigned)(size_t)lp;
  asm volatile("global_load_async_to_lds_b128 %0, %1, off"
               :: "v"(lds_off), "v"(gp) : "memory");
}
__device__ __forceinline__ void wait_async() {
  asm volatile("s_wait_asynccnt 0x0" ::: "memory");
}

// WMMA fragment builders (16x16x32 bf16, wave32)
// A frag: lane row = lane%16; K(e) = e + 16*(lane/16), ap points at row base + kstep
__device__ __forceinline__ v16bf load_a_frag(const __bf16* ap, int half) {
  v16bf a;
#pragma unroll
  for (int e = 0; e < 16; ++e) a[e] = ap[16 * half + e];
  return a;
}
// B frag: lane col = lane%16; K(e) = e + 16*(lane/16); bp points at [n]-row base + kstep + 16*half
__device__ __forceinline__ v16bf load_b_frag(const __bf16* bp) {
  v16bf b;
#pragma unroll
  for (int e = 0; e < 16; ++e) b[e] = bp[e];
  return b;
}

// ---------------------------------------------------------------- embedding
__global__ void embed_kernel(const int* __restrict__ idx, const float* __restrict__ tok,
                             const float* __restrict__ pos, float* __restrict__ x) {
  int t = blockIdx.x;
  int i = threadIdx.x;
  const float4* tp = (const float4*)(tok + (size_t)idx[t] * DMODEL);
  const float4* pp = (const float4*)(pos + (size_t)t * DMODEL);
  float4 a = tp[i], b = pp[i];
  ((float4*)(x + (size_t)t * DMODEL))[i] =
      make_float4(a.x + b.x, a.y + b.y, a.z + b.z, a.w + b.w);
}

// ---------------------------------------------------------------- layernorm -> bf16
__global__ void ln_kernel(const float* __restrict__ x, const float* __restrict__ g,
                          const float* __restrict__ b, __bf16* __restrict__ out) {
  __shared__ float s1[256];
  __shared__ float s2[256];
  int t = blockIdx.x, tid = threadIdx.x;
  const float4* xp = (const float4*)(x + (size_t)t * DMODEL);
  float4 v = xp[tid];
  s1[tid] = v.x + v.y + v.z + v.w;
  s2[tid] = v.x * v.x + v.y * v.y + v.z * v.z + v.w * v.w;
  __syncthreads();
  for (int off = 128; off > 0; off >>= 1) {
    if (tid < off) { s1[tid] += s1[tid + off]; s2[tid] += s2[tid + off]; }
    __syncthreads();
  }
  float mean = s1[0] * (1.0f / DMODEL);
  float var  = s2[0] * (1.0f / DMODEL) - mean * mean;
  float rstd = rsqrtf(var + 1e-5f);
  float4 gv = ((const float4*)g)[tid];
  float4 bv = ((const float4*)b)[tid];
  bf16x4 o;
  o.v[0] = f2bf((v.x - mean) * rstd * gv.x + bv.x);
  o.v[1] = f2bf((v.y - mean) * rstd * gv.y + bv.y);
  o.v[2] = f2bf((v.z - mean) * rstd * gv.z + bv.z);
  o.v[3] = f2bf((v.w - mean) * rstd * gv.w + bv.w);
  ((bf16x4*)(out + (size_t)t * DMODEL))[tid] = o;
}

// ---------------------------------------------------------------- WMMA GEMM
// C[M,N] = A[M,K](bf16) @ B[K,N](f32 weights) + bias (+ Res) (+ ReLU)
// 256 threads (8 wave32); tile 128x64; k-step 32; double-buffered LDS; async A staging.
__device__ __forceinline__ void stage_tiles(const __bf16* __restrict__ A,
                                            const float* __restrict__ B,
                                            int lda, int ldb, int BM, int BN,
                                            int kk, int K, int tid,
                                            __bf16* As, __bf16* Bs) {
#pragma unroll
  for (int j = 0; j < 2; ++j) {
    int idx = tid + 256 * j;
    int row = idx >> 2, c = idx & 3;                 // 4 x 8-elem chunks per 32-k row
    async_copy_b128(A + (size_t)(BM + row) * lda + kk + c * 8, &As[row * 32 + c * 8]);
  }
  int kr = (tid >> 4) * 2;
  int n4 = tid & 15;
  float4 b0 = *(const float4*)(B + (size_t)(kk + kr) * ldb + BN + n4 * 4);
  float4 b1 = *(const float4*)(B + (size_t)(kk + kr + 1) * ldb + BN + n4 * 4);
  float c0[4] = {b0.x, b0.y, b0.z, b0.w};
  float c1[4] = {b1.x, b1.y, b1.z, b1.w};
#pragma unroll
  for (int u = 0; u < 4; ++u) {
    unsigned pk = (unsigned)f2bfu(c0[u]) | ((unsigned)f2bfu(c1[u]) << 16);
    *(unsigned*)&Bs[(n4 * 4 + u) * 32 + kr] = pk;
  }
  if (kk + 32 < K) {
    const float* pf = B + (size_t)(kk + 32 + kr) * ldb + BN + n4 * 4;
    __builtin_prefetch(pf, 0, 1);
  }
}

template <bool RELU, bool HAS_RES, bool OUT_BF16>
__global__ void gemm_wmma_kernel(const __bf16* __restrict__ A, const float* __restrict__ B,
                                 const float* __restrict__ bias, const float* __restrict__ Res,
                                 void* __restrict__ Cv, int M, int N, int K,
                                 int lda, int ldb, int ldc) {
  __shared__ __bf16 As[2][128 * 32];
  __shared__ __bf16 Bs[2][64 * 32];    // [n][k]
  const int tid  = threadIdx.x;
  const int lane = tid & 31;
  const int wave = tid >> 5;
  const int wm   = wave & 3;
  const int wn   = wave >> 2;
  const int BM   = blockIdx.y * 128;
  const int BN   = blockIdx.x * 64;
  const int half = lane >> 4;
  const int l16  = lane & 15;

  v8f acc[2][2];
  acc[0][0] = {}; acc[0][1] = {}; acc[1][0] = {}; acc[1][1] = {};

  stage_tiles(A, B, lda, ldb, BM, BN, 0, K, tid, As[0], Bs[0]);

  int p = 0;
  for (int k0 = 0; k0 < K; k0 += 32) {
    wait_async();
    __syncthreads();
    if (k0 + 32 < K)
      stage_tiles(A, B, lda, ldb, BM, BN, k0 + 32, K, tid, As[p ^ 1], Bs[p ^ 1]);

    v16bf afr[2], bfr[2];
#pragma unroll
    for (int f = 0; f < 2; ++f) {
      afr[f] = load_a_frag(&As[p][(wm * 32 + f * 16 + l16) * 32], half);
      bfr[f] = load_b_frag(&Bs[p][(wn * 32 + f * 16 + l16) * 32 + 16 * half]);
    }
#pragma unroll
    for (int fm = 0; fm < 2; ++fm)
#pragma unroll
      for (int fn = 0; fn < 2; ++fn)
        acc[fm][fn] = __builtin_amdgcn_wmma_f32_16x16x32_bf16(
            false, afr[fm], false, bfr[fn], (short)0, acc[fm][fn], false, false);
    p ^= 1;
  }

#pragma unroll
  for (int fm = 0; fm < 2; ++fm) {
#pragma unroll
    for (int fn = 0; fn < 2; ++fn) {
      int gn = BN + wn * 32 + fn * 16 + l16;
      float bval = bias[gn];
#pragma unroll
      for (int r = 0; r < 8; ++r) {
        int gm = BM + wm * 32 + fm * 16 + half * 8 + r;
        float val = acc[fm][fn][r] + bval;
        if (HAS_RES) val += Res[(size_t)gm * ldc + gn];
        if (RELU) val = fmaxf(val, 0.0f);
        if (OUT_BF16) ((__bf16*)Cv)[(size_t)gm * ldc + gn] = f2bf(val);
        else          ((float*)Cv)[(size_t)gm * ldc + gn] = val;
      }
    }
  }
}

// ---------------------------------------------------------------- WMMA flash attention
// block = (head, 64 queries); key tiles of 64; online softmax; all matmuls on WMMA.
// S = Q.K^T : A = Q [q][d] (LDS natural), B = K^T -> LDS [t][d] natural (async raw copy).
// O += P.V  : A = P [q][t], B = V -> LDS transposed [d][t].
__global__ void attn_wmma_kernel(const __bf16* __restrict__ q, const __bf16* __restrict__ k,
                                 const __bf16* __restrict__ v, __bf16* __restrict__ o) {
  __shared__ __bf16 Qs[64 * 64];
  __shared__ __bf16 Ks[64 * 64];      // [t][d]
  __shared__ __bf16 Vt[64 * 64];      // [d][t]
  __shared__ float  Sb[64 * 64];
  __shared__ __bf16 Pb[64 * 64];      // [q][t]
  __shared__ float  mrow[64], lrow[64], frow[64];
  __shared__ float  red[64 * 4], red2[64 * 4];

  const int tid   = threadIdx.x;
  const int lane  = tid & 31;
  const int wave  = tid >> 5;
  const int half  = lane >> 4;
  const int l16   = lane & 15;
  const int qm    = wave & 3;     // 16-row group of the 64-q tile
  const int wn    = wave >> 2;    // 32-col group (keys in S phase, head-dim in PV phase)
  const int qbase = blockIdx.x * 64;
  const int h     = blockIdx.y;
  const int g     = h >> 2;       // kv group (QPG = 4)
  const int qcol  = h * HEADD;
  const int kcol  = g * HEADD;

  if (tid < 64) { mrow[tid] = -1e30f; lrow[tid] = 0.f; }

  // stage Q tile (64x64 bf16 = 512 x 16B chunks) via async copy
#pragma unroll
  for (int j = 0; j < 2; ++j) {
    int idx = tid + 256 * j;
    int row = idx >> 3, c = idx & 7;
    async_copy_b128(q + (size_t)(qbase + row) * DMODEL + qcol + c * 8, &Qs[row * 64 + c * 8]);
  }

  v8f acc_o[2];
  acc_o[0] = {}; acc_o[1] = {};

  const int ktiles = blockIdx.x + 1;      // causal: keys <= qbase+63
  for (int kt = 0; kt < ktiles; ++kt) {
    // ---- stage K (async raw) and V (transposed) tiles
#pragma unroll
    for (int j = 0; j < 2; ++j) {
      int idx = tid + 256 * j;
      int row = idx >> 3, c = idx & 7;
      async_copy_b128(k + (size_t)(kt * 64 + row) * KVD + kcol + c * 8, &Ks[row * 64 + c * 8]);
    }
#pragma unroll
    for (int j = 0; j < 2; ++j) {
      int idx = tid + 256 * j;
      int row = idx >> 3, c = idx & 7;
      uint4 raw = *(const uint4*)(v + (size_t)(kt * 64 + row) * KVD + kcol + c * 8);
      unsigned w4[4] = {raw.x, raw.y, raw.z, raw.w};
#pragma unroll
      for (int u = 0; u < 4; ++u) {
        Vt[(c * 8 + u * 2) * 64 + row]     = u2bf((unsigned short)(w4[u] & 0xffffu));
        Vt[(c * 8 + u * 2 + 1) * 64 + row] = u2bf((unsigned short)(w4[u] >> 16));
      }
    }
    wait_async();
    __syncthreads();

    // ---- S = Q.K^T (64x64), each wave a 16x32 strip
    v8f sacc[2];
    sacc[0] = {}; sacc[1] = {};
#pragma unroll
    for (int ks = 0; ks < 64; ks += 32) {
      v16bf a = load_a_frag(&Qs[(qm * 16 + l16) * 64 + ks], half);
#pragma unroll
      for (int fn = 0; fn < 2; ++fn) {
        v16bf b = load_b_frag(&Ks[(wn * 32 + fn * 16 + l16) * 64 + ks + 16 * half]);
        sacc[fn] = __builtin_amdgcn_wmma_f32_16x16x32_bf16(
            false, a, false, b, (short)0, sacc[fn], false, false);
      }
    }
#pragma unroll
    for (int fn = 0; fn < 2; ++fn)
#pragma unroll
      for (int r = 0; r < 8; ++r)
        Sb[(qm * 16 + half * 8 + r) * 64 + wn * 32 + fn * 16 + l16] = sacc[fn][r];
    __syncthreads();

    // ---- online softmax: 4 threads per row, 16 cols each
    {
      int ro  = tid & 63;
      int qtr = tid >> 6;
      int ct  = qtr * 16;
      int qg  = qbase + ro;
      float sv[16];
      float lm = -1e30f;
#pragma unroll
      for (int j = 0; j < 16; ++j) {
        int t = kt * 64 + ct + j;
        float s = Sb[ro * 64 + ct + j];
        sv[j] = (t <= qg) ? s * 0.125f : -1e30f;
        lm = fmaxf(lm, sv[j]);
      }
      red[ro * 4 + qtr] = lm;
      __syncthreads();
      float tmax = fmaxf(fmaxf(red[ro * 4 + 0], red[ro * 4 + 1]),
                         fmaxf(red[ro * 4 + 2], red[ro * 4 + 3]));
      float mo = mrow[ro];
      float mn = fmaxf(mo, tmax);
      float ls = 0.f;
#pragma unroll
      for (int j = 0; j < 16; ++j) {
        float p = __expf(sv[j] - mn);
        ls += p;
        Pb[ro * 64 + ct + j] = f2bf(p);
      }
      red2[ro * 4 + qtr] = ls;
      __syncthreads();
      if (qtr == 0) {
        float ts = red2[ro * 4] + red2[ro * 4 + 1] + red2[ro * 4 + 2] + red2[ro * 4 + 3];
        float f = __expf(mo - mn);
        lrow[ro] = lrow[ro] * f + ts;
        frow[ro] = f;
        mrow[ro] = mn;
      }
      __syncthreads();
    }

    // ---- rescale O accumulators, then O += P.V
    {
      float fr[8];
#pragma unroll
      for (int r = 0; r < 8; ++r) fr[r] = frow[qm * 16 + half * 8 + r];
#pragma unroll
      for (int fn = 0; fn < 2; ++fn)
#pragma unroll
        for (int r = 0; r < 8; ++r) acc_o[fn][r] *= fr[r];
    }
#pragma unroll
    for (int ks = 0; ks < 64; ks += 32) {
      v16bf a = load_a_frag(&Pb[(qm * 16 + l16) * 64 + ks], half);
#pragma unroll
      for (int fn = 0; fn < 2; ++fn) {
        v16bf b = load_b_frag(&Vt[(wn * 32 + fn * 16 + l16) * 64 + ks + 16 * half]);
        acc_o[fn] = __builtin_amdgcn_wmma_f32_16x16x32_bf16(
            false, a, false, b, (short)0, acc_o[fn], false, false);
      }
    }
    __syncthreads();   // protect Ks/Vt/Pb/Sb before next tile
  }

  // ---- epilogue: O /= l, write bf16
  {
    float il[8];
#pragma unroll
    for (int r = 0; r < 8; ++r) il[r] = 1.0f / lrow[qm * 16 + half * 8 + r];
#pragma unroll
    for (int fn = 0; fn < 2; ++fn)
#pragma unroll
      for (int r = 0; r < 8; ++r) {
        int row = qm * 16 + half * 8 + r;
        int d   = wn * 32 + fn * 16 + l16;
        o[(size_t)(qbase + row) * DMODEL + qcol + d] = f2bf(acc_o[fn][r] * il[r]);
      }
  }
}

// ---------------------------------------------------------------- host
extern "C" void kernel_launch(void* const* d_in, const int* in_sizes, int n_in,
                              void* d_out, int out_size, void* d_ws, size_t ws_size,
                              hipStream_t stream) {
  (void)in_sizes; (void)n_in; (void)out_size; (void)ws_size;
  const int*   idx = (const int*)  d_in[0];
  // d_in[1] = causal mask (computed analytically in attn kernel)
  const float* tok = (const float*)d_in[2];
  const float* pos = (const float*)d_in[3];
  const float* Wq = (const float*)d_in[4],  *bq  = (const float*)d_in[5];
  const float* Wk = (const float*)d_in[6],  *bk  = (const float*)d_in[7];
  const float* Wv = (const float*)d_in[8],  *bv  = (const float*)d_in[9];
  const float* Wo = (const float*)d_in[10], *bo  = (const float*)d_in[11];
  const float* W1 = (const float*)d_in[12], *b1  = (const float*)d_in[13];
  const float* W2 = (const float*)d_in[14], *b2  = (const float*)d_in[15];
  const float* g1 = (const float*)d_in[16], *be1 = (const float*)d_in[17];
  const float* g2 = (const float*)d_in[18], *be2 = (const float*)d_in[19];
  const float* gf = (const float*)d_in[20], *bef = (const float*)d_in[21];
  const float* Wout = (const float*)d_in[22], *bout = (const float*)d_in[23];

  // workspace: x f32 (4MB) | hbuf bf16 (2MB) | pool (8MB):
  //   pool = qb bf16 (2MB) + kb bf16 (0.5MB) + vb bf16 (0.5MB) + ob bf16 (2MB)
  //   ffn bf16 (8MB) aliases the whole pool (qb/kb/vb/ob dead by then)
  char* base = (char*)d_ws;
  float*  x    = (float*)base;
  __bf16* hb   = (__bf16*)(base + (4 << 20));
  char*   pool = base + (6 << 20);
  __bf16* qb   = (__bf16*)pool;
  __bf16* kb   = (__bf16*)(pool + (2 << 20));
  __bf16* vb   = (__bf16*)(pool + (2 << 20) + (1 << 19));
  __bf16* ob   = (__bf16*)(pool + (3 << 20));
  __bf16* ffn  = (__bf16*)pool;

  dim3 blk(256);
  embed_kernel<<<T_SEQ, blk, 0, stream>>>(idx, tok, pos, x);

  for (int l = 0; l < NLAYER; ++l) {
    const float* Wq_l = Wq + (size_t)l * DMODEL * DMODEL;
    const float* Wk_l = Wk + (size_t)l * DMODEL * KVD;
    const float* Wv_l = Wv + (size_t)l * DMODEL * KVD;
    const float* Wo_l = Wo + (size_t)l * DMODEL * DMODEL;
    const float* W1_l = W1 + (size_t)l * DMODEL * FFDIM;
    const float* W2_l = W2 + (size_t)l * FFDIM * DMODEL;

    ln_kernel<<<T_SEQ, blk, 0, stream>>>(x, g1 + (size_t)l * DMODEL, be1 + (size_t)l * DMODEL, hb);
    gemm_wmma_kernel<false, false, true><<<dim3(DMODEL / 64, T_SEQ / 128), blk, 0, stream>>>(
        hb, Wq_l, bq + (size_t)l * DMODEL, nullptr, qb,
        T_SEQ, DMODEL, DMODEL, DMODEL, DMODEL, DMODEL);
    gemm_wmma_kernel<false, false, true><<<dim3(KVD / 64, T_SEQ / 128), blk, 0, stream>>>(
        hb, Wk_l, bk + (size_t)l * KVD, nullptr, kb,
        T_SEQ, KVD, DMODEL, DMODEL, KVD, KVD);
    gemm_wmma_kernel<false, false, true><<<dim3(KVD / 64, T_SEQ / 128), blk, 0, stream>>>(
        hb, Wv_l, bv + (size_t)l * KVD, nullptr, vb,
        T_SEQ, KVD, DMODEL, DMODEL, KVD, KVD);
    attn_wmma_kernel<<<dim3(T_SEQ / 64, NHEAD), blk, 0, stream>>>(qb, kb, vb, ob);
    gemm_wmma_kernel<false, true, false><<<dim3(DMODEL / 64, T_SEQ / 128), blk, 0, stream>>>(
        ob, Wo_l, bo + (size_t)l * DMODEL, x, x,
        T_SEQ, DMODEL, DMODEL, DMODEL, DMODEL, DMODEL);
    ln_kernel<<<T_SEQ, blk, 0, stream>>>(x, g2 + (size_t)l * DMODEL, be2 + (size_t)l * DMODEL, hb);
    gemm_wmma_kernel<true, false, true><<<dim3(FFDIM / 64, T_SEQ / 128), blk, 0, stream>>>(
        hb, W1_l, b1 + (size_t)l * FFDIM, nullptr, ffn,
        T_SEQ, FFDIM, DMODEL, DMODEL, FFDIM, FFDIM);
    gemm_wmma_kernel<false, true, false><<<dim3(DMODEL / 64, T_SEQ / 128), blk, 0, stream>>>(
        ffn, W2_l, b2 + (size_t)l * DMODEL, x, x,
        T_SEQ, DMODEL, FFDIM, FFDIM, DMODEL, DMODEL);
  }

  ln_kernel<<<T_SEQ, blk, 0, stream>>>(x, gf, bef, hb);
  gemm_wmma_kernel<false, false, false><<<dim3(VOCAB / 64, T_SEQ / 128), blk, 0, stream>>>(
      hb, Wout, bout, nullptr, (float*)d_out,
      T_SEQ, VOCAB, DMODEL, DMODEL, VOCAB, VOCAB);
}
